// NeuralClustering_27427661152816
// MI455X (gfx1250) — matrix operand order repaired and checked by reading, not compiled, not run on hardware
//
#include <hip/hip_runtime.h>
#include <hip/hip_bf16.h>

// ---------------- vector types for WMMA ----------------
typedef __attribute__((ext_vector_type(16))) _Float16 v16h;
typedef __attribute__((ext_vector_type(8)))  _Float16 v8h;
typedef __attribute__((ext_vector_type(8)))  float    v8f;

union V16 { v16h v; v8h h[2]; };

// ---------------- problem constants (from setup_inputs) ----------------
// B=64, N=4096, x_dim=256, h_dim=256, g_dim=512, Hd=512, K=32, n=2048
#define BB    64
#define NN    4096
#define NCUT  2048

// ---------------- workspace layout (bytes, 256-aligned) ----------------
static const size_t OFF_WH1T = 0;                        // [512][256] f16
static const size_t OFF_WH2T = 262144;                   // [256][512] f16
static const size_t OFF_WQ1T = 524288;                   // [512][256] f16
static const size_t OFF_WQ2T = 786432;                   // [256][512] f16
static const size_t OFF_WG1T = 1048576;                  // [512][256] f16
static const size_t OFF_WG2T = 1310720;                  // [512][512] f16
static const size_t OFF_WE1T = 1835008;                  // [512][768] f16
static const size_t OFF_HK   = 2621440;                  // [64][32][256] f32
static const size_t OFF_HN   = 4718592;                  // [64][256] f32
static const size_t OFF_Q    = 4784128;                  // [64][256] f32
static const size_t OFF_GS   = 4849664;                  // [64][65][512] f32
static const size_t OFF_UU   = 13369344;                 // [2112][768] f16

// ---------------- weight transpose + f32->f16 convert ----------------
// dst[o*fin + i] = (f16) src[i*fout + o]
__global__ __launch_bounds__(256) void nc_cvtT_kernel(const float* __restrict__ src,
                                                      _Float16* __restrict__ dst,
                                                      int fin, int fout) {
    int id = blockIdx.x * 256 + threadIdx.x;
    int total = fin * fout;
    if (id >= total) return;
    int i = id % fin;
    int o = id / fin;
    dst[(size_t)o * fin + i] = (_Float16)src[(size_t)i * fout + o];
}

__global__ __launch_bounds__(256) void nc_zero_kernel(float* __restrict__ p, int nfloats) {
    int id = blockIdx.x * 256 + threadIdx.x;
    if (id < nfloats) p[id] = 0.0f;
}

// ---------------- fused 2-layer MLP (256 -> 512 relu -> 256) + reduction ----
// M-tile = 64 rows per workgroup; 8 waves = 4 M-subtiles x 2 N-halves.
// mode 0: t<2048 -> atomic-add into Hk[b][cs[t]][:]; t==2048 -> store hn[b][:]
// mode 1: atomic-add into Q[b][:]
__global__ __launch_bounds__(256) void nc_mlp_sum_kernel(
    const float* __restrict__ data, const int* __restrict__ cs,
    const _Float16* __restrict__ W1T, const float* __restrict__ b1,
    const _Float16* __restrict__ W2T, const float* __restrict__ b2,
    float* __restrict__ Hk, float* __restrict__ hn, float* __restrict__ Q,
    int mode, int tbase, int tcount)
{
    __shared__ _Float16 ldsX[64 * 264] __attribute__((aligned(16)));  // padded stride
    __shared__ _Float16 ldsH[64 * 520] __attribute__((aligned(16)));
    __shared__ int ldsCS[64];

    const int tid  = threadIdx.x;
    const int b    = blockIdx.y;
    const int tile = blockIdx.x;

    // ---- stage X tile (64 rows x 256) into LDS as f16 ----
    {
        const int m  = tid >> 2;        // 0..63
        const int cb = (tid & 3) << 6;  // 0,64,128,192
        const int idx = tile * 64 + m;
        if (idx < tcount) {
            const int t = tbase + idx;
            const float4* src4 = (const float4*)(data + ((size_t)(b * NN + t)) * 256 + cb);
            #pragma unroll
            for (int j = 0; j < 16; ++j) {
                float4 f = src4[j];
                ldsX[m * 264 + cb + j * 4 + 0] = (_Float16)f.x;
                ldsX[m * 264 + cb + j * 4 + 1] = (_Float16)f.y;
                ldsX[m * 264 + cb + j * 4 + 2] = (_Float16)f.z;
                ldsX[m * 264 + cb + j * 4 + 3] = (_Float16)f.w;
            }
        } else {
            #pragma unroll
            for (int j = 0; j < 64; ++j) ldsX[m * 264 + cb + j] = (_Float16)0.0f;
        }
    }
    if (tid < 64) {
        const int idx = tile * 64 + tid;
        const int t = tbase + idx;
        ldsCS[tid] = (mode == 0 && idx < tcount && t < NCUT) ? cs[t] : 0;
    }
    __syncthreads();

    const int w    = tid >> 5;
    const int l    = tid & 31;
    const int hf   = l >> 4;     // lane half
    const int ln   = l & 15;
    const int msub = w & 3;      // M sub-tile (16 rows each)
    const int nh   = w >> 2;     // N half
    const int arow = (msub << 4) + ln;
    const v8f vzero = {};

    // ---- layer 1: H(64x512) = relu(X @ W1 + b1)
    //      wave covers rows [16*msub,16*msub+16), cols [256*nh,256*nh+256) in 2 passes
    for (int p = 0; p < 2; ++p) {
        const int colbase = (nh << 8) + (p << 7);
        v8f acc[8];
        #pragma unroll
        for (int i = 0; i < 8; ++i) acc[i] = vzero;

        for (int kc = 0; kc < 8; ++kc) {
            const int kb = kc * 32;
            V16 a;
            a.h[0] = *(const v8h*)&ldsX[arow * 264 + kb + 16 * hf];
            a.h[1] = *(const v8h*)&ldsX[arow * 264 + kb + 16 * hf + 8];
            #pragma unroll
            for (int ct = 0; ct < 8; ++ct) {
                const int col = colbase + (ct << 4) + ln;
                const v16h bm = *(const v16h*)&W1T[(size_t)col * 256 + kb + 16 * hf];
                acc[ct] = __builtin_amdgcn_wmma_f32_16x16x32_f16(
                    false, a.v, false, bm, (short)0, acc[ct], false, false);
            }
        }
        #pragma unroll
        for (int ct = 0; ct < 8; ++ct) {
            const int col = colbase + (ct << 4) + ln;
            const float bias = b1[col];
            #pragma unroll
            for (int r = 0; r < 8; ++r) {
                float hv = acc[ct][r] + bias;
                hv = hv > 0.0f ? hv : 0.0f;
                ldsH[((msub << 4) + r + 8 * hf) * 520 + col] = (_Float16)hv;
            }
        }
    }
    __syncthreads();

    // ---- layer 2: Y(64x256) = H @ W2 + b2; wave covers 16 rows x 128 cols ----
    {
        const int colbase = nh << 7;
        v8f acc[8];
        #pragma unroll
        for (int i = 0; i < 8; ++i) acc[i] = vzero;

        for (int kc = 0; kc < 16; ++kc) {
            const int kb = kc * 32;
            V16 a;
            a.h[0] = *(const v8h*)&ldsH[arow * 520 + kb + 16 * hf];
            a.h[1] = *(const v8h*)&ldsH[arow * 520 + kb + 16 * hf + 8];
            #pragma unroll
            for (int ct = 0; ct < 8; ++ct) {
                const int col = colbase + (ct << 4) + ln;
                const v16h bm = *(const v16h*)&W2T[(size_t)col * 512 + kb + 16 * hf];
                acc[ct] = __builtin_amdgcn_wmma_f32_16x16x32_f16(
                    false, a.v, false, bm, (short)0, acc[ct], false, false);
            }
        }
        #pragma unroll
        for (int ct = 0; ct < 8; ++ct) {
            const int col = colbase + (ct << 4) + ln;
            const float bias = b2[col];
            #pragma unroll
            for (int r = 0; r < 8; ++r) {
                const int mm  = (msub << 4) + r + 8 * hf;
                const int idx = tile * 64 + mm;
                if (idx >= tcount) continue;
                const int t = tbase + idx;
                const float y = acc[ct][r] + bias;
                if (mode == 0) {
                    if (t < NCUT) {
                        unsafeAtomicAdd(&Hk[((size_t)b * 32 + ldsCS[mm]) * 256 + col], y);
                    } else if (t == NCUT) {
                        hn[b * 256 + col] = y;
                    }
                } else {
                    unsafeAtomicAdd(&Q[b * 256 + col], y);
                }
            }
        }
    }
}

// ---------------- gs = MLP(H, Wg1, bg1, Wg2, bg2); H built on the fly ----------
__global__ __launch_bounds__(256) void nc_gs_kernel(
    const float* __restrict__ Hk, const float* __restrict__ hn,
    const _Float16* __restrict__ W1T, const float* __restrict__ b1,
    const _Float16* __restrict__ W2T, const float* __restrict__ b2,
    float* __restrict__ gs)
{
    __shared__ _Float16 ldsX[16 * 264] __attribute__((aligned(16)));
    __shared__ _Float16 ldsH[16 * 520] __attribute__((aligned(16)));

    const int tid  = threadIdx.x;
    const int tile = blockIdx.x;

    {
        const int m  = tid >> 4;
        const int cb = (tid & 15) << 4;
        const int rr = tile * 16 + m;          // 0..4159
        const int bb = rr / 65;
        const int j  = rr % 65;
        #pragma unroll
        for (int jj = 0; jj < 16; ++jj) {
            const int c = cb + jj;
            float v;
            if (j < 32)      v = Hk[((size_t)bb * 32 + j) * 256 + c];
            else if (j < 64) v = Hk[((size_t)bb * 32 + (j - 32)) * 256 + c] + hn[bb * 256 + c];
            else             v = hn[bb * 256 + c];
            ldsX[m * 264 + c] = (_Float16)v;
        }
    }
    __syncthreads();

    const int w  = tid >> 5;
    const int l  = tid & 31;
    const int hf = l >> 4;
    const int ln = l & 15;
    const v8f vzero = {};

    // layer 1: 256 -> 512, relu
    v8f acc[4];
    #pragma unroll
    for (int i = 0; i < 4; ++i) acc[i] = vzero;
    #pragma unroll
    for (int kc = 0; kc < 8; ++kc) {
        const int kb = kc * 32;
        V16 a;
        a.h[0] = *(const v8h*)&ldsX[ln * 264 + kb + 16 * hf];
        a.h[1] = *(const v8h*)&ldsX[ln * 264 + kb + 16 * hf + 8];
        #pragma unroll
        for (int ct = 0; ct < 4; ++ct) {
            const int col = (w << 6) + (ct << 4) + ln;
            const v16h bm = *(const v16h*)&W1T[(size_t)col * 256 + kb + 16 * hf];
            acc[ct] = __builtin_amdgcn_wmma_f32_16x16x32_f16(
                false, a.v, false, bm, (short)0, acc[ct], false, false);
        }
    }
    #pragma unroll
    for (int ct = 0; ct < 4; ++ct) {
        const int col = (w << 6) + (ct << 4) + ln;
        const float bias = b1[col];
        #pragma unroll
        for (int r = 0; r < 8; ++r) {
            float hv = acc[ct][r] + bias;
            hv = hv > 0.0f ? hv : 0.0f;
            ldsH[(r + 8 * hf) * 520 + col] = (_Float16)hv;
        }
    }
    __syncthreads();

    // layer 2: 512 -> 512
    v8f acc2[4];
    #pragma unroll
    for (int i = 0; i < 4; ++i) acc2[i] = vzero;
    #pragma unroll
    for (int kc = 0; kc < 16; ++kc) {
        const int kb = kc * 32;
        V16 a;
        a.h[0] = *(const v8h*)&ldsH[ln * 520 + kb + 16 * hf];
        a.h[1] = *(const v8h*)&ldsH[ln * 520 + kb + 16 * hf + 8];
        #pragma unroll
        for (int ct = 0; ct < 4; ++ct) {
            const int col = (w << 6) + (ct << 4) + ln;
            const v16h bm = *(const v16h*)&W2T[(size_t)col * 512 + kb + 16 * hf];
            acc2[ct] = __builtin_amdgcn_wmma_f32_16x16x32_f16(
                false, a.v, false, bm, (short)0, acc2[ct], false, false);
        }
    }
    #pragma unroll
    for (int ct = 0; ct < 4; ++ct) {
        const int col = (w << 6) + (ct << 4) + ln;
        const float bias = b2[col];
        #pragma unroll
        for (int r = 0; r < 8; ++r) {
            const int rr = tile * 16 + r + 8 * hf;
            gs[(size_t)rr * 512 + col] = acc2[ct][r] + bias;
        }
    }
}

// ---------------- S/G combine + uu assembly (f16) + G_mask ----------------
__global__ __launch_bounds__(256) void nc_combine_kernel(
    const float* __restrict__ gs, const float* __restrict__ Q,
    _Float16* __restrict__ uu, float* __restrict__ out)
{
    const int b   = blockIdx.x;
    const int tid = threadIdx.x;

    for (int c = tid; c < 512; c += 256) {
        float S = 0.0f;
        for (int k = 0; k < 32; ++k) S += gs[((size_t)b * 65 + k) * 512 + c];
        for (int k = 0; k < 32; ++k) {
            float ge = S - gs[((size_t)b * 65 + k) * 512 + c]
                         + gs[((size_t)b * 65 + 32 + k) * 512 + c];
            uu[((size_t)b * 33 + k) * 768 + c] = (_Float16)ge;
        }
        uu[((size_t)b * 33 + 32) * 768 + c] =
            (_Float16)(S + gs[((size_t)b * 65 + 64) * 512 + c]);
    }
    for (int cq = tid; cq < 256; cq += 256) {
        float qv = Q[b * 256 + cq];
        for (int j = 0; j < 33; ++j)
            uu[((size_t)b * 33 + j) * 768 + 512 + cq] = (_Float16)qv;
    }
    // G_mask = ones (second output, flat-concatenated after E)
    const int gid = b * 256 + tid;
    if (gid < BB * 33) out[BB * 33 + gid] = 1.0f;
}

// ---------------- E = relu(uu @ We1 + be1) @ We2 ----------------
__global__ __launch_bounds__(256) void nc_energy_kernel(
    const _Float16* __restrict__ uu,
    const _Float16* __restrict__ W1T, const float* __restrict__ b1,
    const float* __restrict__ We2, float* __restrict__ out)
{
    __shared__ _Float16 ldsU[16 * 776] __attribute__((aligned(16)));
    __shared__ _Float16 ldsH[16 * 520] __attribute__((aligned(16)));
    __shared__ float    ldsP[256];

    const int tid  = threadIdx.x;
    const int tile = blockIdx.x;

    {
        const int m  = tid >> 4;
        const int cb = (tid & 15) * 48;
        const _Float16* src = uu + ((size_t)(tile * 16 + m)) * 768 + cb;
        #pragma unroll
        for (int j = 0; j < 48; j += 8)
            *(v8h*)&ldsU[m * 776 + cb + j] = *(const v8h*)&src[j];
    }
    __syncthreads();

    const int w  = tid >> 5;
    const int l  = tid & 31;
    const int hf = l >> 4;
    const int ln = l & 15;
    const v8f vzero = {};

    // layer 1: 768 -> 512, relu
    v8f acc[4];
    #pragma unroll
    for (int i = 0; i < 4; ++i) acc[i] = vzero;
    #pragma unroll
    for (int kc = 0; kc < 24; ++kc) {
        const int kb = kc * 32;
        V16 a;
        a.h[0] = *(const v8h*)&ldsU[ln * 776 + kb + 16 * hf];
        a.h[1] = *(const v8h*)&ldsU[ln * 776 + kb + 16 * hf + 8];
        #pragma unroll
        for (int ct = 0; ct < 4; ++ct) {
            const int col = (w << 6) + (ct << 4) + ln;
            const v16h bm = *(const v16h*)&W1T[(size_t)col * 768 + kb + 16 * hf];
            acc[ct] = __builtin_amdgcn_wmma_f32_16x16x32_f16(
                false, a.v, false, bm, (short)0, acc[ct], false, false);
        }
    }
    #pragma unroll
    for (int ct = 0; ct < 4; ++ct) {
        const int col = (w << 6) + (ct << 4) + ln;
        const float bias = b1[col];
        #pragma unroll
        for (int r = 0; r < 8; ++r) {
            float hv = acc[ct][r] + bias;
            hv = hv > 0.0f ? hv : 0.0f;
            ldsH[(r + 8 * hf) * 520 + col] = (_Float16)hv;
        }
    }
    __syncthreads();

    // layer 2: dot(hidden, We2) -> scalar per row (width-1 output; VALU)
    {
        const int r   = tid >> 4;
        const int seg = tid & 15;
        float p = 0.0f;
        #pragma unroll
        for (int j = 0; j < 32; ++j)
            p += (float)ldsH[r * 520 + seg * 32 + j] * We2[seg * 32 + j];
        ldsP[r * 16 + seg] = p;
    }
    __syncthreads();
    if (tid < 16) {
        float s = 0.0f;
        #pragma unroll
        for (int j = 0; j < 16; ++j) s += ldsP[tid * 16 + j];
        out[tile * 16 + tid] = s;
    }
}

// ---------------- host launch ----------------
extern "C" void kernel_launch(void* const* d_in, const int* in_sizes, int n_in,
                              void* d_out, int out_size, void* d_ws, size_t ws_size,
                              hipStream_t stream) {
    const float* data = (const float*)d_in[0];
    const int*   cs   = (const int*)d_in[1];
    const float* Wh1 = (const float*)d_in[3];
    const float* bh1 = (const float*)d_in[4];
    const float* Wh2 = (const float*)d_in[5];
    const float* bh2 = (const float*)d_in[6];
    const float* Wq1 = (const float*)d_in[7];
    const float* bq1 = (const float*)d_in[8];
    const float* Wq2 = (const float*)d_in[9];
    const float* bq2 = (const float*)d_in[10];
    const float* Wg1 = (const float*)d_in[11];
    const float* bg1 = (const float*)d_in[12];
    const float* Wg2 = (const float*)d_in[13];
    const float* bg2 = (const float*)d_in[14];
    const float* We1 = (const float*)d_in[15];
    const float* be1 = (const float*)d_in[16];
    const float* We2 = (const float*)d_in[17];

    char* ws = (char*)d_ws;
    _Float16* WH1T = (_Float16*)(ws + OFF_WH1T);
    _Float16* WH2T = (_Float16*)(ws + OFF_WH2T);
    _Float16* WQ1T = (_Float16*)(ws + OFF_WQ1T);
    _Float16* WQ2T = (_Float16*)(ws + OFF_WQ2T);
    _Float16* WG1T = (_Float16*)(ws + OFF_WG1T);
    _Float16* WG2T = (_Float16*)(ws + OFF_WG2T);
    _Float16* WE1T = (_Float16*)(ws + OFF_WE1T);
    float*    Hk   = (float*)(ws + OFF_HK);
    float*    hn   = (float*)(ws + OFF_HN);
    float*    Q    = (float*)(ws + OFF_Q);
    float*    gsb  = (float*)(ws + OFF_GS);
    _Float16* uu   = (_Float16*)(ws + OFF_UU);
    float*    out  = (float*)d_out;

    // 1) weight transpose/convert to f16 [out][in]
    nc_cvtT_kernel<<<(256 * 512 + 255) / 256, 256, 0, stream>>>(Wh1, WH1T, 256, 512);
    nc_cvtT_kernel<<<(512 * 256 + 255) / 256, 256, 0, stream>>>(Wh2, WH2T, 512, 256);
    nc_cvtT_kernel<<<(256 * 512 + 255) / 256, 256, 0, stream>>>(Wq1, WQ1T, 256, 512);
    nc_cvtT_kernel<<<(512 * 256 + 255) / 256, 256, 0, stream>>>(Wq2, WQ2T, 512, 256);
    nc_cvtT_kernel<<<(256 * 512 + 255) / 256, 256, 0, stream>>>(Wg1, WG1T, 256, 512);
    nc_cvtT_kernel<<<(512 * 512 + 255) / 256, 256, 0, stream>>>(Wg2, WG2T, 512, 512);
    nc_cvtT_kernel<<<(768 * 512 + 255) / 256, 256, 0, stream>>>(We1, WE1T, 768, 512);

    // 2) zero accumulators (Hk + hn + Q contiguous: 557056 floats)
    nc_zero_kernel<<<(557056 + 255) / 256, 256, 0, stream>>>(Hk, 557056);

    // 3) hs path: t in [0, 2048]  (2049 rows/batch -> 33 tiles of 64)
    nc_mlp_sum_kernel<<<dim3(33, BB), 256, 0, stream>>>(
        data, cs, WH1T, bh1, WH2T, bh2, Hk, hn, Q, /*mode=*/0, /*tbase=*/0, /*tcount=*/2049);
    // 4) qs path: t in [2049, 4095]  (2047 rows/batch -> 32 tiles of 64)
    nc_mlp_sum_kernel<<<dim3(32, BB), 256, 0, stream>>>(
        data, cs, WQ1T, bq1, WQ2T, bq2, Hk, hn, Q, /*mode=*/1, /*tbase=*/2049, /*tcount=*/2047);

    // 5) gs = MLP(H)   (B*65 = 4160 rows -> 260 tiles)
    nc_gs_kernel<<<260, 256, 0, stream>>>(Hk, hn, WG1T, bg1, WG2T, bg2, gsb);

    // 6) combine -> uu (f16) + G_mask ones
    nc_combine_kernel<<<BB, 256, 0, stream>>>(gsb, Q, uu, out);

    // 7) E = MLP(uu)   (B*33 = 2112 rows -> 132 tiles)
    nc_energy_kernel<<<132, 256, 0, stream>>>(uu, WE1T, be1, We2, out);
}
